// TriPlaneMulti_44418551775633
// MI455X (gfx1250) — matrix operand order repaired and checked by reading, not compiled, not run on hardware
//
#include <hip/hip_runtime.h>
#include <hip/hip_bf16.h>
#include <math.h>

typedef __attribute__((ext_vector_type(16))) _Float16 v16h;
typedef __attribute__((ext_vector_type(8)))  _Float16 v8h;
typedef __attribute__((ext_vector_type(8)))  float    v8f;
typedef int v4i __attribute__((__vector_size__(16)));   // matches builtin param type

#define P_TOTAL 131072
#define FDIM    64
#define HID     256
#define INDIM   576
#define PTS_PER_BLOCK 64

#define AS1 __attribute__((address_space(1)))
#define AS3 __attribute__((address_space(3)))

// ---------------------------------------------------------------------------
// Async global->LDS copy of 16B, no VGPR staging (tracked by ASYNCcnt).
// ---------------------------------------------------------------------------
__device__ __forceinline__ void async_b128(const _Float16* g, _Float16* l) {
#if __has_builtin(__builtin_amdgcn_global_load_async_to_lds_b128)
    v4i* gv = (v4i*)const_cast<_Float16*>(g);
    __builtin_amdgcn_global_load_async_to_lds_b128(
        (AS1 v4i*)gv, (AS3 v4i*)(v4i*)l, 0, 0);
#else
    *(float4*)l = *(const float4*)g;
#endif
}

__device__ __forceinline__ void async_wait() {
#if __has_builtin(__builtin_amdgcn_s_wait_asynccnt)
    __builtin_amdgcn_s_wait_asynccnt(0);
#else
    asm volatile("s_wait_asynccnt 0" ::: "memory");
#endif
}

// ---------------------------------------------------------------------------
// Kernel 1: plane transpose (C,L,L) f32  ->  (L*L, 64ch) f16, tiled via LDS so
// both the global read and the global write are coalesced.
// ---------------------------------------------------------------------------
__global__ __launch_bounds__(256) void transpose_plane(const float* __restrict__ src,
                                                       _Float16* __restrict__ dst,
                                                       int T /* = L*L */) {
    __shared__ __attribute__((aligned(16))) _Float16 tile[64 * 68];
    const int t0  = blockIdx.x * 64;
    const int tid = threadIdx.x;
#pragma unroll
    for (int i = 0; i < 16; ++i) {
        int idx = i * 256 + tid;         // 0..4095
        int c   = idx >> 6;              // channel 0..63
        int tl  = idx & 63;              // texel in tile
        int t   = t0 + tl;
        float val = (t < T) ? src[(size_t)c * T + t] : 0.f;  // coalesced over tl
        tile[tl * 68 + c] = (_Float16)val;
    }
    __syncthreads();
#pragma unroll
    for (int i = 0; i < 16; ++i) {
        int idx = i * 256 + tid;
        int tl  = idx >> 6;
        int c   = idx & 63;
        int t   = t0 + tl;
        if (t < T) dst[(size_t)t * 64 + c] = tile[tl * 68 + c];  // coalesced over c
    }
}

// ---------------------------------------------------------------------------
// Kernel 2: weight-norm + pack weights into a WMMA B layout whose k order matches load_A.
// B tile is 32x16 f16: lane = (n&15) | ((k&31)>=16 ? 16 : 0), halves j = k&15.
// One block per (padded) output row n; padded rows (n >= Nrows) write zeros.
// ---------------------------------------------------------------------------
__global__ __launch_bounds__(256) void pack_weights(const float* __restrict__ v,
                                                    const float* __restrict__ g,
                                                    _Float16* __restrict__ dst,
                                                    int Nrows, int K, int KT) {
    __shared__ float red[256];
    const int n   = blockIdx.x;
    const int tid = threadIdx.x;
    float s = 0.f;
    if (n < Nrows)
        for (int k = tid; k < K; k += 256) { float t = v[(size_t)n * K + k]; s += t * t; }
    red[tid] = s;
    __syncthreads();
    for (int off = 128; off > 0; off >>= 1) {
        if (tid < off) red[tid] += red[tid + off];
        __syncthreads();
    }
    float scale = 0.f;
    if (n < Nrows) {
        float nrm = sqrtf(red[0]);
        scale = (nrm > 0.f) ? g[n] / nrm : 0.f;
    }
    const int nt = n >> 4, nl = n & 15;
    for (int k = tid; k < K; k += 256) {
        int kt   = k >> 5;
        int kk   = k & 31;
        int lane = nl + ((kk >> 4) << 4);
        int j    = kk & 15;
        float val = (n < Nrows) ? v[(size_t)n * K + k] * scale : 0.f;
        dst[((size_t)(nt * KT + kt) * 32 + lane) * 16 + j] = (_Float16)val;
    }
}

// ---------------------------------------------------------------------------
// Helpers for the fused kernel
// ---------------------------------------------------------------------------
__device__ __forceinline__ v16h load_A(const _Float16* rowbase, int kt, int lhi) {
    // A k order used here (matches the B packing): lane<16 holds K {0..15}, lane>=16 holds
    // K {16..31} of its row -> two contiguous 16B chunks.
    const _Float16* p = rowbase + kt * 32 + lhi * 16;
    v8h lo = *(const v8h*)(p);
    v8h hi = *(const v8h*)(p + 8);
    return __builtin_shufflevector(lo, hi, 0, 1, 2, 3, 4, 5, 6, 7,
                                           8, 9, 10, 11, 12, 13, 14, 15);
}

__device__ __forceinline__ float softplus100(float h) {
    float z = 100.f * h;
    return (fmaxf(z, 0.f) + log1pf(expf(-fabsf(z)))) * 0.01f;
}

// One GEMM layer. A tiles register-resident (reused over all NT n-tiles).
// B n-tiles double-buffer staged into LDS via ASYNC global->LDS DMA, and the
// B operand itself is software-pipelined one K-step ahead in registers so the
// ds_load latency is hidden behind the previous WMMA.
template <int KT, int NT, typename Epi>
__device__ __forceinline__ void layer(const _Float16* __restrict__ arow,
                                      const _Float16* __restrict__ wsrc,
                                      _Float16* __restrict__ bstage,  // [2][KT*512]
                                      int tid, int lane, int lhi, Epi&& epi) {
    constexpr int CH = KT * 64;      // 16B chunks per n-tile (KT*512 halves)
    constexpr int IT = CH / 128;     // chunks per thread (128 threads)

    // hoist this wave's A tiles into registers
    v16h areg[KT];
#pragma unroll
    for (int kt = 0; kt < KT; ++kt) areg[kt] = load_A(arow, kt, lhi);

    // preload n-tile 0 into buffer 0
#pragma unroll
    for (int i = 0; i < IT; ++i) {
        int idx = i * 128 + tid;
        async_b128(wsrc + idx * 8, bstage + idx * 8);
    }
    async_wait();
    __syncthreads();

    for (int nt = 0; nt < NT; ++nt) {
        const int cur = nt & 1;
        if (nt + 1 < NT) {   // kick off async copy of the next n-tile
            const _Float16* s = wsrc + (size_t)(nt + 1) * (KT * 512);
            _Float16* d = bstage + (1 - cur) * (KT * 512);
#pragma unroll
            for (int i = 0; i < IT; ++i) {
                int idx = i * 128 + tid;
                async_b128(s + idx * 8, d + idx * 8);
            }
        }
        const _Float16* bcur = bstage + cur * (KT * 512);
        v8f c = {};
        // software-pipeline B one K-step ahead
        v16h bbuf = *(const v16h*)(bcur + lane * 16);
#pragma unroll
        for (int kt = 0; kt < KT; ++kt) {
            v16h bnext = bbuf;
            if (kt + 1 < KT)
                bnext = *(const v16h*)(bcur + ((kt + 1) * 32 + lane) * 16);
            c = __builtin_amdgcn_wmma_f32_16x16x32_f16(false, areg[kt], false, bbuf,
                                                       (short)0, c, false, false);
            bbuf = bnext;
        }
        epi(nt, c);
        async_wait();      // this wave's async stores to LDS are visible
        __syncthreads();   // swap: all reads of bcur done, other buffer filled
    }
}

// ---------------------------------------------------------------------------
// Kernel 3: fused bilerp-gather + 3-layer MLP. 64 points / block, 4 waves,
// each wave owns a 16-point row tile and runs WMMA f32<-f16 over all N tiles.
// ---------------------------------------------------------------------------
__global__ __launch_bounds__(128) void triplane_mlp(const float* __restrict__ x,
                                                    const _Float16* __restrict__ planes,
                                                    const _Float16* __restrict__ w1p,
                                                    const _Float16* __restrict__ w2p,
                                                    const _Float16* __restrict__ wlp,
                                                    const float* __restrict__ b1,
                                                    const float* __restrict__ b2,
                                                    const float* __restrict__ bl,
                                                    float* __restrict__ out) {
    __shared__ __attribute__((aligned(16))) _Float16 feat[PTS_PER_BLOCK * INDIM]; // 73728 B
    __shared__ __attribute__((aligned(16))) _Float16 hbuf[PTS_PER_BLOCK * HID];   // 32768 B
    __shared__ __attribute__((aligned(16))) _Float16 bstage[2 * 18 * 512];        // 36864 B

    const int tid = threadIdx.x;
    const int p0  = blockIdx.x * PTS_PER_BLOCK;

    // ---- phase 1: bilinear gathers from channel-contiguous f16 planes ----
    const int Ltab[9] = {64, 128, 256, 64, 128, 256, 64, 128, 256};
    const int Otab[9] = {0, 262144, 1310720, 5505024, 5767168, 6815744,
                         11010048, 11272192, 12320768};
    for (int it = 0; it < 36; ++it) {
        int t    = it * 128 + tid;        // 4608 tasks = 64 pts * 9 planes * 8 cgroups
        int cg   = t & 7;
        int rest = t >> 3;
        int pi   = rest % 9;
        int pt   = rest / 9;
        int L    = Ltab[pi];
        const float* xp = x + (size_t)(p0 + pt) * 3;
        float u, v;
        int grp = pi / 3;                 // 0:xy 1:xz 2:yz
        if (grp == 0)      { u = xp[0]; v = xp[1]; }
        else if (grp == 1) { u = xp[0]; v = xp[2]; }
        else               { u = xp[1]; v = xp[2]; }
        float ix = (u + 1.f) * 0.5f * (float)(L - 1);
        float iy = (v + 1.f) * 0.5f * (float)(L - 1);
        float fx = floorf(ix), fy = floorf(iy);
        float wx = ix - fx,    wy = iy - fy;
        int x0i = (int)fx;     x0i = x0i < 0 ? 0 : (x0i > L - 1 ? L - 1 : x0i);
        int x1i = (int)fx + 1; x1i = x1i < 0 ? 0 : (x1i > L - 1 ? L - 1 : x1i);
        int y0i = (int)fy;     y0i = y0i < 0 ? 0 : (y0i > L - 1 ? L - 1 : y0i);
        int y1i = (int)fy + 1; y1i = y1i < 0 ? 0 : (y1i > L - 1 ? L - 1 : y1i);
        const _Float16* base = planes + Otab[pi];
        const v8h nw = *(const v8h*)(base + ((size_t)(y0i * L + x0i)) * 64 + cg * 8);
        const v8h ne = *(const v8h*)(base + ((size_t)(y0i * L + x1i)) * 64 + cg * 8);
        const v8h sw = *(const v8h*)(base + ((size_t)(y1i * L + x0i)) * 64 + cg * 8);
        const v8h se = *(const v8h*)(base + ((size_t)(y1i * L + x1i)) * 64 + cg * 8);
        float wnw = (1.f - wx) * (1.f - wy), wne = wx * (1.f - wy);
        float wsw = (1.f - wx) * wy,         wse = wx * wy;
        _Float16* dst = &feat[pt * INDIM + pi * 64 + cg * 8];
#pragma unroll
        for (int j = 0; j < 8; ++j) {
            float val = wnw * (float)nw[j] + wne * (float)ne[j]
                      + wsw * (float)sw[j] + wse * (float)se[j];
            dst[j] = (_Float16)val;
        }
    }
    __syncthreads();

    const int wave = tid >> 5;
    const int lane = tid & 31;
    const int lrow = lane & 15;   // N column / A row within tile
    const int lhi  = lane >> 4;   // K-half select for A, M-half for C/D

    // ---- layer 1: 576 -> 256, softplus ----
    {
        const _Float16* arow = &feat[(wave * 16 + lrow) * INDIM];
        layer<18, 16>(arow, w1p, bstage, tid, lane, lhi,
            [&](int nt, v8f& c) {
                float bn = b1[nt * 16 + lrow];
#pragma unroll
                for (int r = 0; r < 8; ++r) {
                    float sp = softplus100(c[r] + bn);
                    hbuf[(wave * 16 + lhi * 8 + r) * HID + nt * 16 + lrow] = (_Float16)sp;
                }
            });
    }

    // ---- layer 2: 256 -> 256, softplus; h2 reuses feat storage ----
    _Float16* h2 = feat;  // [64][256]; all layer-1 feat reads completed at the
                          // final barrier inside layer<>()
    {
        const _Float16* arow = &hbuf[(wave * 16 + lrow) * HID];
        layer<8, 16>(arow, w2p, bstage, tid, lane, lhi,
            [&](int nt, v8f& c) {
                float bn = b2[nt * 16 + lrow];
#pragma unroll
                for (int r = 0; r < 8; ++r) {
                    float sp = softplus100(c[r] + bn);
                    h2[(wave * 16 + lhi * 8 + r) * HID + nt * 16 + lrow] = (_Float16)sp;
                }
            });
    }

    // ---- layer 3: 256 -> 65 (padded to 80), linear; masked store ----
    {
        const _Float16* arow = &h2[(wave * 16 + lrow) * HID];
        layer<8, 5>(arow, wlp, bstage, tid, lane, lhi,
            [&](int nt, v8f& c) {
                int n = nt * 16 + lrow;
                float bn = (n < 65) ? bl[n] : 0.f;
#pragma unroll
                for (int r = 0; r < 8; ++r) {
                    float val = c[r] + bn;
                    int prow = p0 + wave * 16 + lhi * 8 + r;
                    if (n < FDIM) {
                        out[(size_t)prow * FDIM + n] = val;        // feat (P,64)
                    } else if (n == FDIM) {
                        out[(size_t)P_TOTAL * FDIM + prow] = val;  // dsdf (P,1)
                    }
                }
            });
    }
}

// ---------------------------------------------------------------------------
// Host launcher
// ---------------------------------------------------------------------------
extern "C" void kernel_launch(void* const* d_in, const int* in_sizes, int n_in,
                              void* d_out, int out_size, void* d_ws, size_t ws_size,
                              hipStream_t stream) {
    (void)in_sizes; (void)n_in; (void)out_size; (void)ws_size;
    const float* x  = (const float*)d_in[0];
    const float* v1 = (const float*)d_in[10];
    const float* g1 = (const float*)d_in[11];
    const float* b1 = (const float*)d_in[12];
    const float* v2 = (const float*)d_in[13];
    const float* g2 = (const float*)d_in[14];
    const float* b2 = (const float*)d_in[15];
    const float* vl = (const float*)d_in[16];
    const float* gl = (const float*)d_in[17];
    const float* bl = (const float*)d_in[18];

    _Float16* ws     = (_Float16*)d_ws;
    _Float16* planes = ws;                     // 16,515,072 halves (33.0 MB)
    _Float16* w1p    = ws + 16515072;          // 147,456 halves
    _Float16* w2p    = w1p + 147456;           // 65,536 halves
    _Float16* wlp    = w2p + 65536;            // 20,480 halves

    const int    Ls[9]   = {64, 128, 256, 64, 128, 256, 64, 128, 256};
    const size_t offs[9] = {0, 262144, 1310720, 5505024, 5767168, 6815744,
                            11010048, 11272192, 12320768};
    for (int i = 0; i < 9; ++i) {
        int T = Ls[i] * Ls[i];
        hipLaunchKernelGGL(transpose_plane, dim3(T / 64), dim3(256), 0, stream,
                           (const float*)d_in[1 + i], planes + offs[i], T);
    }
    hipLaunchKernelGGL(pack_weights, dim3(256), dim3(256), 0, stream, v1, g1, w1p, 256, 576, 18);
    hipLaunchKernelGGL(pack_weights, dim3(256), dim3(256), 0, stream, v2, g2, w2p, 256, 256, 8);
    hipLaunchKernelGGL(pack_weights, dim3(80),  dim3(256), 0, stream, vl, gl, wlp, 65, 256, 8);

    hipLaunchKernelGGL(triplane_mlp, dim3(P_TOTAL / PTS_PER_BLOCK), dim3(128), 0, stream,
                       x, planes, w1p, w2p, wlp, b1, b2, bl, (float*)d_out);
}
